// NeuralRadianceField_9062380994654
// MI455X (gfx1250) — matrix operand order repaired and checked by reading, not compiled, not run on hardware
//
#include <hip/hip_runtime.h>

typedef __attribute__((ext_vector_type(16))) _Float16     v16h;
typedef __attribute__((ext_vector_type(8)))  float        v8f;
typedef __attribute__((ext_vector_type(4)))  unsigned int v4u;

namespace {
constexpr int KSTRIDE = 328;   // halves per LDS activation row (mult of 8 -> 16B aligned rows)

// workspace offsets (in halves) for f16 zero-padded weights, layout [Opad][Kpad] row-major
constexpr int OFF_WX0   = 0;                      // 256 x 64   (63 -> 64)
constexpr int OFF_WX1   = OFF_WX0   + 256 * 64;   // 256 x 256
constexpr int OFF_WX2   = OFF_WX1   + 256 * 256;
constexpr int OFF_WX3   = OFF_WX2   + 256 * 256;
constexpr int OFF_WX4   = OFF_WX3   + 256 * 256;  // 256 x 320  (319 -> 320)
constexpr int OFF_WX5   = OFF_WX4   + 256 * 320;
constexpr int OFF_WX6   = OFF_WX5   + 256 * 256;
constexpr int OFF_WX7   = OFF_WX6   + 256 * 256;
constexpr int OFF_WDEN  = OFF_WX7   + 256 * 256;  // 16 x 256   (1 -> 16 rows)
constexpr int OFF_WFEAT = OFF_WDEN  + 16 * 256;   // 256 x 256
constexpr int OFF_WD0   = OFF_WFEAT + 256 * 256;  // 128 x 288  (283 -> 288)
constexpr int OFF_WRGB  = OFF_WD0   + 128 * 288;  // 16 x 128   (3 -> 16 rows)
} // namespace

__device__ __forceinline__ v16h frag_from(v4u a, v4u b) {
  union { v4u u[2]; v16h h; } t;
  t.u[0] = a; t.u[1] = b;
  return t.h;
}

__device__ __forceinline__ v8f wmma_f16(v16h a, v16h b, v8f c) {
  // (neg_a, A, neg_b, B, c_mod, C, reuse_a, reuse_b)
  return __builtin_amdgcn_wmma_f32_16x16x32_f16(false, a, false, b, (short)0, c,
                                                false, false);
}

// X[64][0..16*nColTiles) = relu(X[64][0..K) @ W^T + b), written back in place.
// W is f16 row-major [16*nColTiles][K] (K padded to mult of 32).
// Job mapping: wave owns col-tiles {wave, wave+8}; iterates all 4 row-tiles.
// -> each weight tile is fetched exactly once per workgroup; A frags shared.
__device__ __forceinline__ void dense_layer(const _Float16* __restrict__ W,
                                            const float* __restrict__ bias,
                                            _Float16* X, int K, int nColTiles) {
  const int lane = threadIdx.x & 31;
  const int wave = threadIdx.x >> 5;
  const int n    = lane & 15;    // A row / B,C,D column within tile
  const int hi   = lane >> 4;    // half-wave selector

  v8f acc[8];
#pragma unroll
  for (int i = 0; i < 8; ++i) acc[i] = (v8f){0.f, 0.f, 0.f, 0.f, 0.f, 0.f, 0.f, 0.f};

  for (int k0 = 0; k0 < K; k0 += 32) {
    // A 16x32 f16 frags for all 4 row-tiles:
    // lanes0-15 K=[0..8)+[16..24), lanes16-31 K=[8..16)+[24..32); row m = n
    v16h a[4];
#pragma unroll
    for (int rt = 0; rt < 4; ++rt) {
      const _Float16* xrow = X + (rt * 16 + n) * KSTRIDE + k0;
      a[rt] = frag_from(*(const v4u*)(xrow + 8 * hi),
                        *(const v4u*)(xrow + 16 + 8 * hi));
    }
#pragma unroll
    for (int c = 0; c < 2; ++c) {
      if (c == 1 && nColTiles < 16) break;
      int ct = wave + 8 * c;
      // B 32x16 f16: lanes0-15 K=[0..8)+[16..24), lanes16-31 K=[8..16)+[24..32); column n
      const _Float16* wrow = W + (ct * 16 + n) * K + k0 + 8 * hi;
      v16h b = frag_from(*(const v4u*)(wrow), *(const v4u*)(wrow + 16));
#pragma unroll
      for (int rt = 0; rt < 4; ++rt)
        acc[c * 4 + rt] = wmma_f16(a[rt], b, acc[c * 4 + rt]);
    }
  }
  __syncthreads();  // all reads of X done before overwrite
#pragma unroll
  for (int c = 0; c < 2; ++c) {
    if (c == 1 && nColTiles < 16) break;
    int ct = wave + 8 * c;
    float bn = bias[ct * 16 + n];
#pragma unroll
    for (int rt = 0; rt < 4; ++rt) {
#pragma unroll
      for (int r = 0; r < 8; ++r) {
        float v = acc[c * 4 + rt][r] + bn;
        v = fmaxf(v, 0.0f);
        // C/D layout: VGPR r -> M = r + 8*hi, N = n
        X[(rt * 16 + r + 8 * hi) * KSTRIDE + ct * 16 + n] = (_Float16)v;
      }
    }
  }
  __syncthreads();
}

// Small heads (O padded to 16): density (relu, component 0) or rgb (sigmoid, components 1..3)
__device__ __forceinline__ void head_layer(const _Float16* __restrict__ W,
                                           const float* __restrict__ bias,
                                           const _Float16* X, int K,
                                           float* __restrict__ out, int rowBase,
                                           bool sigmoid_rgb) {
  const int lane = threadIdx.x & 31;
  const int wave = threadIdx.x >> 5;
  if (wave >= 4) return;                 // wave-uniform: EXEC all-ones inside
  const int n = lane & 15, hi = lane >> 4;
  const int rt = wave;
  v8f c = {0.f, 0.f, 0.f, 0.f, 0.f, 0.f, 0.f, 0.f};
  const _Float16* xrow = X + (rt * 16 + n) * KSTRIDE;
  const _Float16* wrow = W + n * K;
  for (int k0 = 0; k0 < K; k0 += 32) {
    v4u a0 = *(const v4u*)(xrow + k0 + 8 * hi);
    v4u a1 = *(const v4u*)(xrow + k0 + 16 + 8 * hi);
    v4u b0 = *(const v4u*)(wrow + k0 + 8 * hi);
    v4u b1 = *(const v4u*)(wrow + k0 + 16 + 8 * hi);
    c = wmma_f16(frag_from(a0, a1), frag_from(b0, b1), c);
  }
  if (!sigmoid_rgb) {
    if (n == 0) {
      float bn = bias[0];
#pragma unroll
      for (int r = 0; r < 8; ++r) {
        float v = fmaxf(c[r] + bn, 0.0f);
        out[(rowBase + rt * 16 + r + 8 * hi) * 4 + 0] = v;
      }
    }
  } else {
    if (n < 3) {
      float bn = bias[n];
#pragma unroll
      for (int r = 0; r < 8; ++r) {
        float v = c[r] + bn;
        out[(rowBase + rt * 16 + r + 8 * hi) * 4 + 1 + n] = 1.0f / (1.0f + __expf(-v));
      }
    }
  }
}

struct Biases {
  const float* bx[8];
  const float* bd0;
  const float* bden;
  const float* bfeat;
  const float* brgb;
};

__global__ __launch_bounds__(256)
void nerf_fused_kernel(const float* __restrict__ pts, const float* __restrict__ dirs,
                       const _Float16* __restrict__ ws, Biases B,
                       float* __restrict__ out) {
  __shared__ _Float16 X[64 * KSTRIDE];  // activations: cols 0..255 hidden, 256..319 embeddings
  __shared__ float    P[64 * 3];
  const int tile = blockIdx.x;          // one ray (64 samples) per workgroup
  const int tid  = threadIdx.x;

  for (int e = tid; e < 64 * 3; e += 256) P[e] = pts[tile * (64 * 3) + e];
  __syncthreads();

  // xyz harmonic embedding: [sin(x*2^j) x30, cos x30, x x3, pad0] -> cols 0..63 AND 256..319
  for (int e = tid; e < 64 * 64; e += 256) {
    int i = e >> 6, c = e & 63;
    float v;
    if (c < 30)      { int cc = c;      v = __sinf(P[i * 3 + cc / 10] * (float)(1 << (cc % 10))); }
    else if (c < 60) { int cc = c - 30; v = __cosf(P[i * 3 + cc / 10] * (float)(1 << (cc % 10))); }
    else if (c < 63) v = P[i * 3 + (c - 60)];
    else             v = 0.0f;
    _Float16 h = (_Float16)v;
    X[i * KSTRIDE + c]       = h;
    X[i * KSTRIDE + 256 + c] = h;
  }
  __syncthreads();

  dense_layer(ws + OFF_WX0, B.bx[0], X, 64, 16);
  dense_layer(ws + OFF_WX1, B.bx[1], X, 256, 16);
  dense_layer(ws + OFF_WX2, B.bx[2], X, 256, 16);
  dense_layer(ws + OFF_WX3, B.bx[3], X, 256, 16);
  dense_layer(ws + OFF_WX4, B.bx[4], X, 320, 16);   // skip concat: cols 256..319 = emb
  dense_layer(ws + OFF_WX5, B.bx[5], X, 256, 16);
  dense_layer(ws + OFF_WX6, B.bx[6], X, 256, 16);
  dense_layer(ws + OFF_WX7, B.bx[7], X, 256, 16);

  head_layer(ws + OFF_WDEN, B.bden, X, 256, out, tile * 64, /*sigmoid_rgb=*/false);
  dense_layer(ws + OFF_WFEAT, B.bfeat, X, 256, 16);

  // direction harmonic embedding (shared by all 64 samples of the ray) -> cols 256..287
  for (int e = tid; e < 64 * 32; e += 256) {
    int i = e >> 5, c = e & 31;
    float v;
    if (c < 12)      { int cc = c;      v = __sinf(dirs[tile * 3 + cc / 4] * (float)(1 << (cc & 3))); }
    else if (c < 24) { int cc = c - 12; v = __cosf(dirs[tile * 3 + cc / 4] * (float)(1 << (cc & 3))); }
    else if (c < 27) v = dirs[tile * 3 + (c - 24)];
    else             v = 0.0f;
    X[i * KSTRIDE + 256 + c] = (_Float16)v;
  }
  __syncthreads();

  dense_layer(ws + OFF_WD0, B.bd0, X, 288, 8);      // -> cols 0..127
  head_layer(ws + OFF_WRGB, B.brgb, X, 128, out, tile * 64, /*sigmoid_rgb=*/true);
}

// f32 -> f16 weight conversion with zero padding: dst [Opad][Kpad], src [O][K]
__global__ void cvt_pad_kernel(const float* __restrict__ src, _Float16* __restrict__ dst,
                               int O, int K, int Kpad, int total) {
  int idx = blockIdx.x * 256 + threadIdx.x;
  if (idx >= total) return;
  int o = idx / Kpad, k = idx - o * Kpad;
  float v = (o < O && k < K) ? src[o * K + k] : 0.0f;
  dst[idx] = (_Float16)v;
}

extern "C" void kernel_launch(void* const* d_in, const int* in_sizes, int n_in,
                              void* d_out, int out_size, void* d_ws, size_t ws_size,
                              hipStream_t stream) {
  (void)in_sizes; (void)n_in; (void)out_size; (void)ws_size;
  const float* pts  = (const float*)d_in[0];
  const float* dirs = (const float*)d_in[1];
  const float* Wx[8];
  Biases B;
  for (int i = 0; i < 8; ++i) {
    Wx[i]   = (const float*)d_in[2 + 2 * i];
    B.bx[i] = (const float*)d_in[3 + 2 * i];
  }
  const float* Wd0  = (const float*)d_in[18]; B.bd0  = (const float*)d_in[19];
  const float* Wden = (const float*)d_in[20]; B.bden = (const float*)d_in[21];
  const float* Wfeat= (const float*)d_in[22]; B.bfeat= (const float*)d_in[23];
  const float* Wrgb = (const float*)d_in[24]; B.brgb = (const float*)d_in[25];

  _Float16* ws = (_Float16*)d_ws;
  float*   out = (float*)d_out;

  auto cvt = [&](const float* src, int off, int O, int K, int Opad, int Kpad) {
    int total = Opad * Kpad;
    cvt_pad_kernel<<<(total + 255) / 256, 256, 0, stream>>>(src, ws + off, O, K, Kpad, total);
  };
  cvt(Wx[0], OFF_WX0, 256, 63, 256, 64);
  cvt(Wx[1], OFF_WX1, 256, 256, 256, 256);
  cvt(Wx[2], OFF_WX2, 256, 256, 256, 256);
  cvt(Wx[3], OFF_WX3, 256, 256, 256, 256);
  cvt(Wx[4], OFF_WX4, 256, 319, 256, 320);
  cvt(Wx[5], OFF_WX5, 256, 256, 256, 256);
  cvt(Wx[6], OFF_WX6, 256, 256, 256, 256);
  cvt(Wx[7], OFF_WX7, 256, 256, 256, 256);
  cvt(Wden,  OFF_WDEN,  1,  256, 16, 256);
  cvt(Wfeat, OFF_WFEAT, 256, 256, 256, 256);
  cvt(Wd0,   OFF_WD0,   128, 283, 128, 288);
  cvt(Wrgb,  OFF_WRGB,  3,  128, 16, 128);

  // 4096 rays, one ray (64 samples) per workgroup of 8 wave32's
  nerf_fused_kernel<<<4096, 256, 0, stream>>>(pts, dirs, ws, B, out);
}
